// NeuralPointNet_28080496181701
// MI455X (gfx1250) — matrix operand order repaired and checked by reading, not compiled, not run on hardware
//
#include <hip/hip_runtime.h>
#include <hip/hip_bf16.h>
#include <float.h>

// ---------------------------------------------------------------------------
// PointNet++ (MSG) forward for MI455X / gfx1250.
// Dense MLP layers -> wave32 WMMA GEMM (bf16 in, f32 accumulate):
//     v_wmma_f32_16x16x32_bf16
// Weights are pre-transposed/padded to bf16 so the GEMM inner loop has NO
// bounds checks: contiguous b128 loads only.
// ---------------------------------------------------------------------------

typedef __attribute__((ext_vector_type(16))) __bf16 bf16x16;
typedef __attribute__((ext_vector_type(8)))  float  floatx8;
typedef __attribute__((ext_vector_type(4)))  float  f32x4;

#define WPB 4  // waves per block in the GEMM kernel (128 threads)

static __host__ __device__ inline int pad32(int x) { return (x + 31) & ~31; }

// ---------------------------------------------------------------------------
// Weight prep: W[K,N] f32 (row-major) -> Wt[Np,Kp] bf16 (transposed, padded).
// Wt[n*Kp + k] = bf16(W[k*N + n]), zero outside [K)x[N).
// ---------------------------------------------------------------------------
__global__ void prep_weight_kernel(const float* __restrict__ W,
                                   __bf16* __restrict__ Wt,
                                   int K, int Ncol, int Kp, int Np)
{
    long t = (long)blockIdx.x * blockDim.x + threadIdx.x;
    long total = (long)Np * Kp;
    if (t >= total) return;
    int k = (int)(t % Kp);
    int n = (int)(t / Kp);
    float v = (k < K && n < Ncol) ? W[(long)k * Ncol + n] : 0.0f;
    Wt[t] = (__bf16)v;
}

// ---------------------------------------------------------------------------
// Fused linear (+BN+ReLU) GEMM.  Y[M,ldY] = act(BN(X[M,Kp] * Wt^T + b))
// One wave owns a 16x16 tile.  Fragment element mappings (wave32; A/B use this code's contiguous k order, not the ISA table order):
//   A 16x32 bf16 : lane l -> row M = l&15, half = l>>4
//                  element e -> K = e + 16*half
//                  => per lane: one contiguous 16-float run (k0+16h)
//   B 32x16 bf16 : lane l -> col N = l&15, half = l>>4
//                  element e -> K = e + 16*half
//                  => per lane: one contiguous 16-bf16 run of Wt row
//   C/D 16x16 f32: element r -> row M = r + 8*half, col N = l&15
// X rows have stride Kp (=pad32(K)) with a zero-filled tail, so the loop body
// is branch-free.  Rows >= M are clamped (memory-safe, results discarded).
// ---------------------------------------------------------------------------
__global__ void wmma_linear_bn_relu(const float* __restrict__ X,
                                    const __bf16* __restrict__ Wt,
                                    const float* __restrict__ bias,
                                    const float* __restrict__ gamma,
                                    const float* __restrict__ beta,
                                    const float* __restrict__ mean,
                                    const float* __restrict__ var,
                                    float* __restrict__ Y,
                                    int M, int Kp, int Ncol, int ldY, int fuse)
{
    const int lane = threadIdx.x & 31;
    const int wave = threadIdx.x >> 5;
    const int tilesM = (M + 15) >> 4;
    const int tilesN = ldY >> 4;
    const long total = (long)tilesM * tilesN;
    const long gw = (long)blockIdx.x * WPB + wave;
    if (gw >= total) return;                 // wave-uniform: EXEC stays all-1s
    const int tm = (int)(gw % tilesM);
    const int tn = (int)(gw / tilesM);
    const int row0 = tm << 4, col0 = tn << 4;
    const int hl  = lane >> 4;               // lane half (0 or 1)
    const int l16 = lane & 15;

    floatx8 acc = {0.f, 0.f, 0.f, 0.f, 0.f, 0.f, 0.f, 0.f};

    int arow = row0 + l16;
    if (arow >= M) arow = M - 1;             // clamp: memory-safe, not stored
    const int col = col0 + l16;              // always < ldY (<= Np of Wt)
    const float*  Xr = X  + (long)arow * Kp;
    const __bf16* Wr = Wt + (long)col  * Kp;

#pragma unroll 2
    for (int k0 = 0; k0 < Kp; k0 += 32) {
        const int abase = k0 + (hl << 4);
        f32x4 x0 = *(const f32x4*)(Xr + abase);
        f32x4 x1 = *(const f32x4*)(Xr + abase + 4);
        f32x4 x2 = *(const f32x4*)(Xr + abase + 8);
        f32x4 x3 = *(const f32x4*)(Xr + abase + 12);
        bf16x16 a;
#pragma unroll
        for (int e = 0; e < 4; ++e) {
            a[e]      = (__bf16)x0[e];
            a[e + 4]  = (__bf16)x1[e];
            a[e + 8]  = (__bf16)x2[e];
            a[e + 12] = (__bf16)x3[e];
        }
        bf16x16 b = *(const bf16x16*)(Wr + k0 + (hl << 4));
        acc = __builtin_amdgcn_wmma_f32_16x16x32_bf16(
                  /*neg_a=*/false, a, /*neg_b=*/false, b,
                  /*c_mod=*/(short)0, acc, /*reuse_a=*/false, /*reuse_b=*/false);
    }

    // -------- epilogue (pad columns write zeros so next layer's A is clean)
    const int cc = (col < Ncol) ? col : (Ncol - 1);
    float bb = bias ? bias[cc] : 0.0f;
    float g = 1.f, bt = 0.f, mn = 0.f, iv = 1.f;
    if (fuse) {
        g  = gamma[cc];
        bt = beta[cc];
        mn = mean[cc];
        iv = rsqrtf(var[cc] + 1e-5f);
    }
#pragma unroll
    for (int r = 0; r < 8; ++r) {
        int row = row0 + r + (hl << 3);
        if (row < M) {
            float v = acc[r] + bb;
            if (fuse) { v = g * (v - mn) * iv + bt; v = fmaxf(v, 0.0f); }
            if (col >= Ncol) v = 0.0f;
            Y[(long)row * ldY + col] = v;
        }
    }
}

// ---------------------------------------------------------------------------
// Farthest point sampling + gather of sampled coordinates.
// One block (256 threads) per batch; min-dists live in registers.
// ---------------------------------------------------------------------------
__global__ void fps_gather_kernel(const float* __restrict__ xyz,
                                  float* __restrict__ new_xyz,
                                  int N, int S)
{
    const int b = blockIdx.x;
    const int tid = threadIdx.x;              // 256 threads
    const float* P = xyz + (long)b * N * 3;
    const int npt = (N + 255) / 256;          // <= 16 for N<=4096
    float dmin[16];
    for (int t = 0; t < npt; ++t) dmin[t] = 1e10f;

    __shared__ float s_val[256];
    __shared__ int   s_idx[256];
    __shared__ float s_c[3];

    int far = 0;
    for (int it = 0; it < S; ++it) {
        if (tid == 0) {
            float cx = P[(long)far * 3 + 0];
            float cy = P[(long)far * 3 + 1];
            float cz = P[(long)far * 3 + 2];
            s_c[0] = cx; s_c[1] = cy; s_c[2] = cz;
            new_xyz[((long)b * S + it) * 3 + 0] = cx;
            new_xyz[((long)b * S + it) * 3 + 1] = cy;
            new_xyz[((long)b * S + it) * 3 + 2] = cz;
        }
        __syncthreads();
        float cx = s_c[0], cy = s_c[1], cz = s_c[2];
        float best = -1.0f; int bi = 0;
        for (int t = 0; t < npt; ++t) {
            int n = tid + t * 256;
            if (n < N) {
                float dx = P[(long)n * 3 + 0] - cx;
                float dy = P[(long)n * 3 + 1] - cy;
                float dz = P[(long)n * 3 + 2] - cz;
                float d = dx * dx + dy * dy + dz * dz;
                float dm = fminf(dmin[t], d);
                dmin[t] = dm;
                if (dm > best) { best = dm; bi = n; }
            }
        }
        s_val[tid] = best; s_idx[tid] = bi;
        __syncthreads();
        for (int off = 128; off > 0; off >>= 1) {
            if (tid < off) {
                if (s_val[tid + off] > s_val[tid] ||
                    (s_val[tid + off] == s_val[tid] && s_idx[tid + off] < s_idx[tid])) {
                    s_val[tid] = s_val[tid + off];
                    s_idx[tid] = s_idx[tid + off];
                }
            }
            __syncthreads();
        }
        far = s_idx[0];
        __syncthreads();
    }
}

// ---------------------------------------------------------------------------
// Ball query: first `ns` ascending indices with d2 <= r2, padded with the
// first hit (or N-1 if none), matching the reference's sort/clip logic.
// ---------------------------------------------------------------------------
__global__ void ball_query_kernel(const float* __restrict__ xyz,
                                  const float* __restrict__ nxyz,
                                  int* __restrict__ idx,
                                  int Bn, int N, int S, int ns, float r2)
{
    int t = blockIdx.x * blockDim.x + threadIdx.x;
    if (t >= Bn * S) return;
    int b = t / S;
    const float* q = nxyz + (long)t * 3;
    float qx = q[0], qy = q[1], qz = q[2];
    const float* P = xyz + (long)b * N * 3;
    int* o = idx + (long)t * ns;
    int cnt = 0, first = N - 1;
    for (int n = 0; n < N; ++n) {
        float dx = P[(long)n * 3 + 0] - qx;
        float dy = P[(long)n * 3 + 1] - qy;
        float dz = P[(long)n * 3 + 2] - qz;
        float d = dx * dx + dy * dy + dz * dz;
        if (d <= r2) {
            if (cnt == 0) first = n;
            o[cnt++] = n;
            if (cnt == ns) break;
        }
    }
    for (int c = cnt; c < ns; ++c) o[c] = first;
}

// ---------------------------------------------------------------------------
// Group: out row = [ xyz[idx]-new_xyz , feats[idx] , 0-pad to ld ]
// ---------------------------------------------------------------------------
__global__ void group_kernel(const float* __restrict__ xyz,
                             const float* __restrict__ nxyz,
                             const float* __restrict__ feats,
                             const int* __restrict__ idx,
                             float* __restrict__ out,
                             int Bn, int N, int S, int ns, int Cf, int ld)
{
    long t = (long)blockIdx.x * blockDim.x + threadIdx.x;
    long total = (long)Bn * S * ns;
    if (t >= total) return;
    int n = idx[t];
    long bi = t / ns;                 // b*S + i
    int b = (int)(bi / S);
    const float* p = xyz + ((long)b * N + n) * 3;
    const float* q = nxyz + bi * 3;
    float* o = out + t * ld;
    o[0] = p[0] - q[0];
    o[1] = p[1] - q[1];
    o[2] = p[2] - q[2];
    if (feats) {
        const float* f = feats + ((long)b * N + n) * Cf;
        for (int c = 0; c < Cf; ++c) o[3 + c] = f[c];
    }
    for (int c = 3 + Cf; c < ld; ++c) o[c] = 0.0f;
}

// ---------------------------------------------------------------------------
// Max-pool over sample axis (input stride ldh), write to channel slice of the
// concatenated feature buffer (stride Ctot).
// ---------------------------------------------------------------------------
__global__ void maxpool_concat_kernel(const float* __restrict__ h,
                                      float* __restrict__ out,
                                      int Bn, int S, int ns, int C, int ldh,
                                      int Ctot, int coff)
{
    long t = (long)blockIdx.x * blockDim.x + threadIdx.x;
    long total = (long)Bn * S * C;
    if (t >= total) return;
    int c = (int)(t % C);
    long bi = t / C;                  // b*S + i
    const float* hp = h + (bi * ns) * (long)ldh + c;
    float m = -FLT_MAX;
    for (int s = 0; s < ns; ++s) m = fmaxf(m, hp[(long)s * ldh]);
    out[bi * Ctot + coff + c] = m;
}

// ---------------------------------------------------------------------------
// 3-NN search + inverse-distance weights.
// ---------------------------------------------------------------------------
__global__ void three_nn_kernel(const float* __restrict__ unk,
                                const float* __restrict__ kn,
                                int* __restrict__ idx3,
                                float* __restrict__ w3,
                                int Bn, int Su, int Sk)
{
    int t = blockIdx.x * blockDim.x + threadIdx.x;
    if (t >= Bn * Su) return;
    int b = t / Su;
    const float* q = unk + (long)t * 3;
    float qx = q[0], qy = q[1], qz = q[2];
    const float* P = kn + (long)b * Sk * 3;
    float d0 = FLT_MAX, d1 = FLT_MAX, d2 = FLT_MAX;
    int i0 = 0, i1 = 0, i2 = 0;
    for (int n = 0; n < Sk; ++n) {
        float dx = P[(long)n * 3 + 0] - qx;
        float dy = P[(long)n * 3 + 1] - qy;
        float dz = P[(long)n * 3 + 2] - qz;
        float d = dx * dx + dy * dy + dz * dz;
        if (d < d0)      { d2 = d1; i2 = i1; d1 = d0; i1 = i0; d0 = d; i0 = n; }
        else if (d < d1) { d2 = d1; i2 = i1; d1 = d;  i1 = n; }
        else if (d < d2) { d2 = d;  i2 = n; }
    }
    float w0 = 1.0f / (d0 + 1e-8f);
    float w1 = 1.0f / (d1 + 1e-8f);
    float w2 = 1.0f / (d2 + 1e-8f);
    float s = w0 + w1 + w2;
    w0 /= s; w1 /= s; w2 /= s;
    idx3[(long)t * 3 + 0] = i0; idx3[(long)t * 3 + 1] = i1; idx3[(long)t * 3 + 2] = i2;
    w3[(long)t * 3 + 0]  = w0; w3[(long)t * 3 + 1]  = w1; w3[(long)t * 3 + 2]  = w2;
}

// ---------------------------------------------------------------------------
// Interp + concat skip:  out[b,u,:] = [ sum_j w_j*kf[idx_j] , uf , 0-pad ]
// Output rows have stride ld (= pad32(Ck+Cu)).
// ---------------------------------------------------------------------------
__global__ void interp_concat_kernel(const float* __restrict__ kf,
                                     const float* __restrict__ uf,
                                     const int* __restrict__ idx3,
                                     const float* __restrict__ w3,
                                     float* __restrict__ out,
                                     int Bn, int Su, int Sk, int Ck, int Cu,
                                     int ld)
{
    long t = (long)blockIdx.x * blockDim.x + threadIdx.x;
    long total = (long)Bn * Su * ld;
    if (t >= total) return;
    int c = (int)(t % ld);
    long bu = t / ld;                 // b*Su + u
    int b = (int)(bu / Su);
    float v = 0.0f;
    if (c < Ck) {
        const int*   id = idx3 + bu * 3;
        const float* w  = w3 + bu * 3;
        const float* kb = kf + (long)b * Sk * Ck;
        v = w[0] * kb[(long)id[0] * Ck + c]
          + w[1] * kb[(long)id[1] * Ck + c]
          + w[2] * kb[(long)id[2] * Ck + c];
    } else if (c < Ck + Cu) {
        v = uf[bu * (long)Cu + (c - Ck)];
    }
    out[t] = v;
}

// ---------------------------------------------------------------------------
// (B,N,C) row-stride ldx -> (B,C,N) final transpose.
// ---------------------------------------------------------------------------
__global__ void transpose_out_kernel(const float* __restrict__ x,
                                     float* __restrict__ out,
                                     int Bn, int N, int C, int ldx)
{
    long t = (long)blockIdx.x * blockDim.x + threadIdx.x;
    long total = (long)Bn * N * C;
    if (t >= total) return;
    int c = (int)(t % C);
    long bn = t / C;
    int n = (int)(bn % N);
    int b = (int)(bn / N);
    out[((long)b * C + c) * (long)N + n] = x[bn * (long)ldx + c];
}

// ---------------------------------------------------------------------------
// Host orchestration.
// Param flattening assumption: JAX pytree order — dict keys sorted
// ('fc1','fc2','fp','sa'; per-layer 'W','b','beta','gamma','mean','var'),
// lists in order.  d_in[0] = input_data, then 200 parameter arrays.
// ---------------------------------------------------------------------------
extern "C" void kernel_launch(void* const* d_in, const int* in_sizes, int n_in,
                              void* d_out, int out_size, void* d_ws, size_t ws_size,
                              hipStream_t stream)
{
    (void)in_sizes; (void)n_in; (void)out_size; (void)ws_size;
    const int B = 8;
    static const int   NS[5]          = {4096, 1024, 256, 64, 16};
    static const float RAD[4][2]      = {{0.1f,0.5f},{0.5f,1.0f},{1.0f,2.0f},{2.0f,4.0f}};
    static const int   NSAMP[4][2]    = {{16,32},{16,32},{16,32},{16,32}};
    static const int   SA_CH[4][2][4] = {
        {{3,16,16,32},   {3,32,32,64}},
        {{99,64,64,128}, {99,64,96,128}},
        {{259,128,196,256},{259,128,196,256}},
        {{515,256,256,512},{515,256,384,512}}};
    static const int   FP_CH[4][3]    = {{1536,512,512},{768,512,512},{608,256,256},{256,128,128}};

    const float* input = (const float*)d_in[0];

    struct Layer { const float *W, *b, *beta, *gamma, *mean, *var; };
    int cur = 1;
    auto nextp = [&]() { return (const float*)d_in[cur++]; };
    auto nextL = [&]() {
        Layer L;
        L.W = nextp(); L.b = nextp(); L.beta = nextp();
        L.gamma = nextp(); L.mean = nextp(); L.var = nextp();
        return L;
    };

    Layer fc1L = nextL();
    Layer fc2L; fc2L.W = nextp(); fc2L.b = nextp();
    fc2L.beta = fc2L.gamma = fc2L.mean = fc2L.var = nullptr;
    Layer fpL[4][2];
    for (int i = 0; i < 4; ++i)
        for (int l = 0; l < 2; ++l) fpL[i][l] = nextL();
    Layer saL[4][2][3];
    for (int i = 0; i < 4; ++i)
        for (int j = 0; j < 2; ++j)
            for (int l = 0; l < 3; ++l) saL[i][j][l] = nextL();

    // --- workspace bump allocator (256B aligned) ---
    char* wp = (char*)d_ws;
    auto allocBytes = [&](long nb) { void* p = (void*)wp; wp += ((nb + 255) / 256) * 256; return p; };
    auto allocF = [&](long n) { return (float*)allocBytes(n * 4); };
    auto allocI = [&](long n) { return (int*)allocBytes(n * 4); };

    auto blocks = [](long n) { return dim3((unsigned)((n + 255) / 256)); };

    // --- prepped layers: transposed/padded bf16 weights ---
    struct PL { const float *bias, *gamma, *beta, *mean, *var;
                const __bf16* Wt; int K, N, Kp, Np; };
    auto prep = [&](const Layer& L, int K, int N) {
        PL p;
        p.K = K; p.N = N; p.Kp = pad32(K); p.Np = pad32(N);
        __bf16* wt = (__bf16*)allocBytes((long)p.Np * p.Kp * 2);
        prep_weight_kernel<<<blocks((long)p.Np * p.Kp), dim3(256), 0, stream>>>(
            L.W, wt, K, N, p.Kp, p.Np);
        p.Wt = wt;
        p.bias = L.b; p.gamma = L.gamma; p.beta = L.beta; p.mean = L.mean; p.var = L.var;
        return p;
    };

    PL sa[4][2][3];
    for (int i = 0; i < 4; ++i)
        for (int j = 0; j < 2; ++j)
            for (int l = 0; l < 3; ++l)
                sa[i][j][l] = prep(saL[i][j][l], SA_CH[i][j][l], SA_CH[i][j][l + 1]);
    PL fp[4][2];
    for (int i = 0; i < 4; ++i)
        for (int l = 0; l < 2; ++l)
            fp[i][l] = prep(fpL[i][l], FP_CH[i][l], FP_CH[i][l + 1]);
    PL fc1 = prep(fc1L, 128, 64);
    PL fc2 = prep(fc2L, 64, 8);

    auto runLinear = [&](const float* X, const PL& p, float* Y, long M, int fuse) {
        long tiles = (long)((M + 15) / 16) * (p.Np / 16);
        unsigned grid = (unsigned)((tiles + WPB - 1) / WPB);
        wmma_linear_bn_relu<<<dim3(grid), dim3(32 * WPB), 0, stream>>>(
            X, p.Wt, p.bias, p.gamma, p.beta, p.mean, p.var,
            Y, (int)M, p.Kp, p.N, p.Np, fuse);
    };

    // --- activation buffers ---
    const float* lx[5];  lx[0] = input;
    float* lxw[5];
    for (int i = 0; i < 4; ++i) lxw[i + 1] = allocF((long)B * NS[i + 1] * 3);
    float* lf[5];  lf[0] = nullptr;
    int lc[5];     lc[0] = 0;
    for (int i = 0; i < 4; ++i)
        lf[i + 1] = allocF((long)B * NS[i + 1] * (SA_CH[i][0][3] + SA_CH[i][1][3]));
    float* lfn[4];
    for (int i = 0; i < 4; ++i) lfn[3 - i] = allocF((long)B * NS[3 - i] * FP_CH[i][2]);
    float* gbuf = allocF(9L * 1024 * 1024);
    float* ping = allocF(9L * 1024 * 1024);
    float* pong = allocF(9L * 1024 * 1024);
    int*   idxb = allocI((long)B * 1024 * 32);
    int*   idx3 = allocI((long)B * 4096 * 3);
    float* w3   = allocF((long)B * 4096 * 3);

    // ---------------- Set abstraction levels ----------------
    for (int i = 0; i < 4; ++i) {
        int N = NS[i], S = NS[i + 1];
        fps_gather_kernel<<<dim3(B), dim3(256), 0, stream>>>(lx[i], lxw[i + 1], N, S);
        lx[i + 1] = lxw[i + 1];
        int Ctot = SA_CH[i][0][3] + SA_CH[i][1][3];
        for (int j = 0; j < 2; ++j) {
            int ns = NSAMP[i][j];
            float r = RAD[i][j];
            ball_query_kernel<<<blocks((long)B * S), dim3(256), 0, stream>>>(
                lx[i], lx[i + 1], idxb, B, N, S, ns, r * r);
            long Mg = (long)B * S * ns;
            group_kernel<<<blocks(Mg), dim3(256), 0, stream>>>(
                lx[i], lx[i + 1], lf[i], idxb, gbuf, B, N, S, ns, lc[i],
                sa[i][j][0].Kp);
            const float* cin = gbuf;
            for (int l = 0; l < 3; ++l) {
                float* outp = (l & 1) ? pong : ping;
                runLinear(cin, sa[i][j][l], outp, Mg, 1);
                cin = outp;
            }
            int Cl = sa[i][j][2].N, ldh = sa[i][j][2].Np;
            int off = (j == 0) ? 0 : SA_CH[i][0][3];
            maxpool_concat_kernel<<<blocks((long)B * S * Cl), dim3(256), 0, stream>>>(
                cin, lf[i + 1], B, S, ns, Cl, ldh, Ctot, off);
        }
        lc[i + 1] = Ctot;
    }

    // ---------------- Feature propagation levels ----------------
    const float* lfc[5];
    for (int i = 0; i < 5; ++i) lfc[i] = lf[i];
    for (int i = 0; i < 4; ++i) {
        int li = 3 - i;
        int Su = NS[li], Sk = NS[li + 1];
        three_nn_kernel<<<blocks((long)B * Su), dim3(256), 0, stream>>>(
            lx[li], lx[li + 1], idx3, w3, B, Su, Sk);
        int Ck = lc[li + 1], Cu = lc[li];
        int ldf = fp[i][0].Kp;                      // pad32(Ck+Cu)
        interp_concat_kernel<<<blocks((long)B * Su * ldf), dim3(256), 0, stream>>>(
            lfc[li + 1], lfc[li], idx3, w3, gbuf, B, Su, Sk, Ck, Cu, ldf);
        const float* cin = gbuf;
        for (int l = 0; l < 2; ++l) {
            float* outp = (l == 1) ? lfn[li] : ping;
            runLinear(cin, fp[i][l], outp, (long)B * Su, 1);
            cin = outp;
        }
        lfc[li] = lfn[li];
        lc[li] = fp[i][1].N;                        // 512/512/256/128, all %32==0
    }

    // ---------------- Head ----------------
    runLinear(lfc[0], fc1, ping, (long)B * NS[0], 1);     // 128 -> 64 (+BN+ReLU)
    runLinear(ping, fc2, pong, (long)B * NS[0], 0);       // 64 -> 8  (bias only)
    transpose_out_kernel<<<blocks((long)B * NS[0] * 8), dim3(256), 0, stream>>>(
        pong, (float*)d_out, B, NS[0], 8, fc2.Np);
}
